// GPTOSSTransformerStack_36197984371236
// MI455X (gfx1250) — matrix operand order repaired and checked by reading, not compiled, not run on hardware
//
#include <hip/hip_runtime.h>
#include <hip/hip_bf16.h>
#include <math.h>

// Model constants (from reference)
#define LNUM   4
#define HDIM   1024
#define NHEADS 16
#define NKVH   4
#define HEADD  64
#define FFDIM  2048
#define ENUM   8
#define BB     2
#define SS     2048
#define TOK    (BB*SS)

typedef __attribute__((ext_vector_type(16))) __bf16 bf16x16;
typedef __attribute__((ext_vector_type(8)))  float  v8f;

union Frag16 { bf16x16 v; unsigned u[8]; uint4 q[2]; };

// native f32->bf16 (RNE) via hardware convert
__device__ __forceinline__ unsigned short f2bf(float f) {
  union { __bf16 b; unsigned short u; } c;
  c.b = (__bf16)f;
  return c.u;
}
// packed pair: low short = lo, high short = hi  (lowers to v_cvt_pk_bf16_f32)
__device__ __forceinline__ unsigned pk2(float lo, float hi) {
  union { __bf16 b[2]; unsigned u; } c;
  c.b[0] = (__bf16)lo; c.b[1] = (__bf16)hi;
  return c.u;
}

// ---------------------------------------------------------------------------
// RMSNorm: one block (256 threads) per token
// ---------------------------------------------------------------------------
__global__ __launch_bounds__(256) void rmsnorm_kernel(
    const float* __restrict__ x, const float* __restrict__ w,
    float* __restrict__ y)
{
  __shared__ float red[8];
  const int t = blockIdx.x, tid = threadIdx.x;
  const float* xr = x + (size_t)t * HDIM;
  float ss = 0.f;
  for (int i = tid; i < HDIM; i += 256) { float v = xr[i]; ss += v * v; }
  for (int o = 1; o < 32; o <<= 1) ss += __shfl_xor(ss, o, 32);
  if ((tid & 31) == 0) red[tid >> 5] = ss;
  __syncthreads();
  if (tid < 32) {
    float v = (tid < 8) ? red[tid] : 0.f;
    for (int o = 1; o < 8; o <<= 1) v += __shfl_xor(v, o, 32);
    if (tid == 0) red[0] = v;
  }
  __syncthreads();
  const float inv = rsqrtf(red[0] / (float)HDIM + 1e-5f);
  float* yr = y + (size_t)t * HDIM;
  for (int i = tid; i < HDIM; i += 256) yr[i] = xr[i] * inv * w[i];
}

// ---------------------------------------------------------------------------
// WMMA GEMM, 128x128 block tile, 8 waves as 4(M) x 2(N), wave tile 32x64.
// Double-buffered LDS; global loads of tile k+1 issued before WMMAs on tile k.
// B is staged in 2(k)x4(n) sub-blocks so v_cvt_pk_bf16_f32 directly yields the
// packed (k,k+1) dword of the fragment layout -> single ds_store_b32 per col.
//   gateOther -> v = silu(gateOther[m,n]) * v    (SwiGLU fuse)
//   rowScale  -> v *= rowScale[m*rsStride+rsOff] (top-k combine)
//   accumulate -> C += v else C = v
// ---------------------------------------------------------------------------
__global__ __launch_bounds__(256) void gemm_wmma_kernel(
    const float* __restrict__ A, int lda,
    const float* __restrict__ Bw, int ldb,
    float* __restrict__ C, int ldc,
    int M, int N, int Kd,
    const float* __restrict__ rowScale, int rsStride, int rsOff,
    const float* __restrict__ gateOther, int accumulate)
{
  // rows padded to 40 shorts = 80B (16B multiple) so uint4 reads are aligned
  __shared__ __align__(16) unsigned short As[2][128][40];   // [m][k]
  __shared__ __align__(16) unsigned short Bs[2][128][40];   // [n][k] transposed
  const int tid  = threadIdx.x;
  const int wave = tid >> 5, lane = tid & 31;
  const int half = lane >> 4, l16 = lane & 15;
  const int wm = wave & 3, wn = wave >> 2;               // 4 x 2 wave grid
  const int bm = blockIdx.x * 128, bn = blockIdx.y * 128;

  v8f acc[2][4] = {};
  float4 fa[4], fb0[2], fb1[2];

  // prologue: load + stage tile 0
#pragma unroll
  for (int t = 0; t < 4; ++t) {
    int idx = tid + t * 256;
    fa[t] = *(const float4*)(A + (size_t)(bm + (idx >> 3)) * lda + ((idx & 7) << 2));
  }
#pragma unroll
  for (int t = 0; t < 2; ++t) {
    int idx = tid + t * 256;
    int kr2 = (idx >> 5) << 1, n4 = (idx & 31) << 2;
    fb0[t] = *(const float4*)(Bw + (size_t)kr2 * ldb + bn + n4);
    fb1[t] = *(const float4*)(Bw + (size_t)(kr2 + 1) * ldb + bn + n4);
  }
#pragma unroll
  for (int t = 0; t < 4; ++t) {
    int idx = tid + t * 256;
    int row = idx >> 3, ca = (idx & 7) << 2;
    uint2 d; d.x = pk2(fa[t].x, fa[t].y); d.y = pk2(fa[t].z, fa[t].w);
    *(uint2*)&As[0][row][ca] = d;
  }
#pragma unroll
  for (int t = 0; t < 2; ++t) {
    int idx = tid + t * 256;
    int kr2 = (idx >> 5) << 1, n4 = (idx & 31) << 2;
    *(unsigned*)&Bs[0][n4 + 0][kr2] = pk2(fb0[t].x, fb1[t].x);
    *(unsigned*)&Bs[0][n4 + 1][kr2] = pk2(fb0[t].y, fb1[t].y);
    *(unsigned*)&Bs[0][n4 + 2][kr2] = pk2(fb0[t].z, fb1[t].z);
    *(unsigned*)&Bs[0][n4 + 3][kr2] = pk2(fb0[t].w, fb1[t].w);
  }
  __syncthreads();

  const int nk = Kd >> 5;
  for (int kt = 0; kt < nk; ++kt) {
    const int cur = kt & 1;
    // issue global loads for next tile before computing on current one
    if (kt + 1 < nk) {
      const int k0 = (kt + 1) << 5;
#pragma unroll
      for (int t = 0; t < 4; ++t) {
        int idx = tid + t * 256;
        fa[t] = *(const float4*)(A + (size_t)(bm + (idx >> 3)) * lda + k0 + ((idx & 7) << 2));
      }
#pragma unroll
      for (int t = 0; t < 2; ++t) {
        int idx = tid + t * 256;
        int kr2 = (idx >> 5) << 1, n4 = (idx & 31) << 2;
        fb0[t] = *(const float4*)(Bw + (size_t)(k0 + kr2) * ldb + bn + n4);
        fb1[t] = *(const float4*)(Bw + (size_t)(k0 + kr2 + 1) * ldb + bn + n4);
      }
      if (kt + 2 < nk) {
        __builtin_prefetch(A + (size_t)(bm + (tid >> 1)) * lda + ((kt + 2) << 5) + (tid & 1) * 16, 0, 1);
        __builtin_prefetch(Bw + (size_t)(((kt + 2) << 5) + (tid & 31)) * ldb + bn + (tid >> 5) * 16, 0, 1);
      }
    }
    // A fragments: two contiguous 16B runs per lane -> 2x ds_load_b128 each
    Frag16 a[2];
#pragma unroll
    for (int i = 0; i < 2; ++i) {
      const uint4* p = (const uint4*)&As[cur][wm * 32 + i * 16 + l16][0];
      a[i].q[0] = p[half];
      a[i].q[1] = p[2 + half];
    }
#pragma unroll
    for (int t = 0; t < 4; ++t) {
      Frag16 b;                   // two contiguous 16B runs -> 2x ds_load_b128
      const uint4* p = (const uint4*)&Bs[cur][wn * 64 + t * 16 + l16][0];
      b.q[0] = p[half];
      b.q[1] = p[2 + half];
      acc[0][t] = __builtin_amdgcn_wmma_f32_16x16x32_bf16(
          false, a[0].v, false, b.v, (short)0, acc[0][t], false, false);
      acc[1][t] = __builtin_amdgcn_wmma_f32_16x16x32_bf16(
          false, a[1].v, false, b.v, (short)0, acc[1][t], false, false);
    }
    // stage next tile into the other buffer
    if (kt + 1 < nk) {
      const int nxt = cur ^ 1;
#pragma unroll
      for (int t = 0; t < 4; ++t) {
        int idx = tid + t * 256;
        int row = idx >> 3, ca = (idx & 7) << 2;
        uint2 d; d.x = pk2(fa[t].x, fa[t].y); d.y = pk2(fa[t].z, fa[t].w);
        *(uint2*)&As[nxt][row][ca] = d;
      }
#pragma unroll
      for (int t = 0; t < 2; ++t) {
        int idx = tid + t * 256;
        int kr2 = (idx >> 5) << 1, n4 = (idx & 31) << 2;
        *(unsigned*)&Bs[nxt][n4 + 0][kr2] = pk2(fb0[t].x, fb1[t].x);
        *(unsigned*)&Bs[nxt][n4 + 1][kr2] = pk2(fb0[t].y, fb1[t].y);
        *(unsigned*)&Bs[nxt][n4 + 2][kr2] = pk2(fb0[t].z, fb1[t].z);
        *(unsigned*)&Bs[nxt][n4 + 3][kr2] = pk2(fb0[t].w, fb1[t].w);
      }
    }
    __syncthreads();
  }

  // epilogue: C layout vgpr r -> M = r + 8*half, N = l16
#pragma unroll
  for (int i = 0; i < 2; ++i) {
#pragma unroll
    for (int t = 0; t < 4; ++t) {
#pragma unroll
      for (int r = 0; r < 8; ++r) {
        int gm = bm + wm * 32 + i * 16 + r + 8 * half;
        int gn = bn + wn * 64 + t * 16 + l16;
        if (gm >= M || gn >= N) continue;
        float vv = acc[i][t][r];
        size_t ci = (size_t)gm * ldc + gn;
        if (gateOther) { float g = gateOther[ci]; vv *= g / (1.f + __expf(-g)); }
        if (rowScale)  vv *= rowScale[(size_t)gm * rsStride + rsOff];
        if (accumulate) C[ci] += vv; else C[ci] = vv;
      }
    }
  }
}

// ---------------------------------------------------------------------------
// RoPE in place on q[T,NH,HD] and k[T,NKV,HD]; block = 32 lanes = HD/2 pairs
// ---------------------------------------------------------------------------
__global__ void rope_kernel(float* __restrict__ Qd, float* __restrict__ Kd)
{
  const int t   = blockIdx.x;            // token
  const int hh  = blockIdx.y;            // 0..NH+NKV-1
  const int pos = t % SS;
  const int lane = threadIdx.x;          // 0..31
  float* ptr = (hh < NHEADS)
      ? (Qd + (size_t)t * NHEADS * HEADD + hh * HEADD)
      : (Kd + (size_t)t * NKVH   * HEADD + (hh - NHEADS) * HEADD);
  const float inv = __expf(-(2.0f * lane / (float)HEADD) * 9.210340371976184f);
  const float ang = (float)pos * inv;
  float s, c;
  __sincosf(ang, &s, &c);
  const float x1 = ptr[lane], x2 = ptr[lane + 32];
  ptr[lane]      = x1 * c - x2 * s;
  ptr[lane + 32] = x2 * c + x1 * s;
}

// ---------------------------------------------------------------------------
// Router: one wave per token -> comb[t,E] = normalized top-2 softmax weights
// ---------------------------------------------------------------------------
__global__ void router_kernel(const float* __restrict__ xn,
                              const float* __restrict__ wr,
                              float* __restrict__ comb)
{
  const int t = blockIdx.x, lane = threadIdx.x;
  float lg[ENUM];
#pragma unroll
  for (int e = 0; e < ENUM; ++e) {
    float p = 0.f;
    for (int hI = lane; hI < HDIM; hI += 32)
      p += xn[(size_t)t * HDIM + hI] * wr[(size_t)hI * ENUM + e];
    for (int o = 1; o < 32; o <<= 1) p += __shfl_xor(p, o, 32);
    lg[e] = p;
  }
  float mx = lg[0];
#pragma unroll
  for (int e = 1; e < ENUM; ++e) mx = fmaxf(mx, lg[e]);
  float pe[ENUM], sum = 0.f;
#pragma unroll
  for (int e = 0; e < ENUM; ++e) { pe[e] = __expf(lg[e] - mx); sum += pe[e]; }
#pragma unroll
  for (int e = 0; e < ENUM; ++e) pe[e] /= sum;
  int i1 = 0; float v1 = pe[0];
#pragma unroll
  for (int e = 1; e < ENUM; ++e) if (pe[e] > v1) { v1 = pe[e]; i1 = e; }
  int i2 = -1; float v2 = -1.f;
#pragma unroll
  for (int e = 0; e < ENUM; ++e) if (e != i1 && pe[e] > v2) { v2 = pe[e]; i2 = e; }
  const float s2 = v1 + v2;
  if (lane < ENUM)
    comb[(size_t)t * ENUM + lane] =
        (lane == i1) ? (v1 / s2) : ((lane == i2) ? (v2 / s2) : 0.f);
}

// ---------------------------------------------------------------------------
// Residual add + clip
// ---------------------------------------------------------------------------
__global__ void resid_clip_kernel(const float* __restrict__ res,
                                  const float* __restrict__ hv,
                                  float* __restrict__ out, int n)
{
  int i = blockIdx.x * 256 + threadIdx.x;
  if (i < n) {
    float v = res[i] + hv[i];
    out[i] = fminf(100.f, fmaxf(-100.f, v));
  }
}

// ---------------------------------------------------------------------------
// Flash attention, causal, GQA. 8 waves x 16 q-rows per block, 32-key tiles.
// Q*K^T and P*V via v_wmma_f32_16x16x32_bf16; online softmax in registers.
// All LDS fragment reads are ds_load_b128; V staged as packed key-pairs.
// ---------------------------------------------------------------------------
__global__ __launch_bounds__(256) void attn_kernel(
    const float* __restrict__ Q, const float* __restrict__ Kg,
    const float* __restrict__ Vg, float* __restrict__ O)
{
  __shared__ __align__(16) unsigned short Ks[32][72];       // [key][hd], 144B rows
  __shared__ __align__(16) unsigned short Vs[64][40];       // [hd-col][key], 80B rows
  __shared__ __align__(16) unsigned short Ps[8][16][40];    // per-wave P tiles
  const int tid  = threadIdx.x;
  const int wave = tid >> 5, lane = tid & 31;
  const int half = lane >> 4, l16 = lane & 15;
  const int bh = blockIdx.y, b = bh / NHEADS, h = bh % NHEADS;
  const int kvh = h / (NHEADS / NKVH);
  const int qb = blockIdx.x * 128;
  const int wbase = qb + wave * 16;

  // Q fragments (RoPE already applied): per-lane A layout = two 8-float runs
  Frag16 qa[2];
  {
    const float* qptr = Q + ((size_t)(b * SS + wbase + l16) * NHEADS + h) * HEADD;
#pragma unroll
    for (int c = 0; c < 2; ++c) {
      const float4 f0 = *(const float4*)(qptr + c * 32 + half * 8);
      const float4 f1 = *(const float4*)(qptr + c * 32 + half * 8 + 4);
      const float4 f2 = *(const float4*)(qptr + c * 32 + 16 + half * 8);
      const float4 f3 = *(const float4*)(qptr + c * 32 + 16 + half * 8 + 4);
      qa[c].u[0] = pk2(f0.x, f0.y); qa[c].u[1] = pk2(f0.z, f0.w);
      qa[c].u[2] = pk2(f1.x, f1.y); qa[c].u[3] = pk2(f1.z, f1.w);
      qa[c].u[4] = pk2(f2.x, f2.y); qa[c].u[5] = pk2(f2.z, f2.w);
      qa[c].u[6] = pk2(f3.x, f3.y); qa[c].u[7] = pk2(f3.z, f3.w);
    }
  }

  v8f oacc[4] = {};
  float m_i[8], l_i[8];
#pragma unroll
  for (int r = 0; r < 8; ++r) { m_i[r] = -1e30f; l_i[r] = 0.f; }

  const int nkb = qb / 32 + 4;                 // causal: keys <= qb+127
  for (int kbi = 0; kbi < nkb; ++kbi) {
    const int kb = kbi * 32;
    // stage K (row-major, packed pairs along hd) and V (transposed, packed
    // key-pairs so cvt_pk output is stored directly with ds_store_b32)
    float4 fk[2];
#pragma unroll
    for (int t = 0; t < 2; ++t) {
      int idx = tid + t * 256;
      int key = idx >> 4, h4 = (idx & 15) << 2;
      fk[t] = *(const float4*)(Kg + ((size_t)(b * SS + kb + key) * NKVH + kvh) * HEADD + h4);
    }
    float4 fv0, fv1;
    {
      int kr2 = (tid >> 4) << 1, h4 = (tid & 15) << 2;
      size_t base = ((size_t)(b * SS + kb + kr2) * NKVH + kvh) * HEADD + h4;
      fv0 = *(const float4*)(Vg + base);
      fv1 = *(const float4*)(Vg + base + (size_t)NKVH * HEADD);
    }
#pragma unroll
    for (int t = 0; t < 2; ++t) {
      int idx = tid + t * 256;
      int key = idx >> 4, h4 = (idx & 15) << 2;
      uint2 d; d.x = pk2(fk[t].x, fk[t].y); d.y = pk2(fk[t].z, fk[t].w);
      *(uint2*)&Ks[key][h4] = d;
    }
    {
      int kr2 = (tid >> 4) << 1, h4 = (tid & 15) << 2;
      *(unsigned*)&Vs[h4 + 0][kr2] = pk2(fv0.x, fv1.x);
      *(unsigned*)&Vs[h4 + 1][kr2] = pk2(fv0.y, fv1.y);
      *(unsigned*)&Vs[h4 + 2][kr2] = pk2(fv0.z, fv1.z);
      *(unsigned*)&Vs[h4 + 3][kr2] = pk2(fv0.w, fv1.w);
    }
    __syncthreads();

    if (kb <= wbase + 15) {                    // wave has unmasked work
      // scores 16x32 = two 16x16 C tiles, each accumulated over 2 hd-chunks
      v8f st[2] = {};
#pragma unroll
      for (int t = 0; t < 2; ++t) {
        const uint4* p = (const uint4*)&Ks[t * 16 + l16][0];
#pragma unroll
        for (int c = 0; c < 2; ++c) {
          Frag16 kbf;
          kbf.q[0] = p[c * 4 + half];
          kbf.q[1] = p[c * 4 + 2 + half];
          st[t] = __builtin_amdgcn_wmma_f32_16x16x32_bf16(
              false, qa[c].v, false, kbf.v, (short)0, st[t], false, false);
        }
      }
      // online softmax (row = r + 8*half within wave tile)
      float p0[8], p1[8], corr[8];
#pragma unroll
      for (int r = 0; r < 8; ++r) {
        int qrow = wbase + r + 8 * half;
        float s0 = (kb + l16      <= qrow) ? st[0][r] * 0.125f : -1e30f;
        float s1 = (kb + 16 + l16 <= qrow) ? st[1][r] * 0.125f : -1e30f;
        float mx = fmaxf(s0, s1);
        for (int o = 1; o < 16; o <<= 1) mx = fmaxf(mx, __shfl_xor(mx, o, 32));
        float mn = fmaxf(m_i[r], mx);
        corr[r] = __expf(m_i[r] - mn);
        m_i[r] = mn;
        float e0 = __expf(s0 - mn), e1 = __expf(s1 - mn);
        float rs = e0 + e1;
        for (int o = 1; o < 16; o <<= 1) rs += __shfl_xor(rs, o, 32);
        l_i[r] = l_i[r] * corr[r] + rs;
        p0[r] = e0; p1[r] = e1;
      }
      // P tile C-layout -> A-layout via wave-private LDS
#pragma unroll
      for (int r = 0; r < 8; ++r) {
        int m = r + 8 * half;
        Ps[wave][m][l16]      = f2bf(p0[r]);
        Ps[wave][m][16 + l16] = f2bf(p1[r]);
      }
      __builtin_amdgcn_wave_barrier();
      Frag16 pa;
      {
        const uint4* p = (const uint4*)&Ps[wave][l16][0];
        pa.q[0] = p[half];
        pa.q[1] = p[2 + half];
      }
      // O = O*corr + P@V  (4 column tiles of 16)
#pragma unroll
      for (int t = 0; t < 4; ++t) {
#pragma unroll
        for (int r = 0; r < 8; ++r) oacc[t][r] *= corr[r];
        Frag16 vb;
        const uint4* p = (const uint4*)&Vs[t * 16 + l16][0];
        vb.q[0] = p[half];
        vb.q[1] = p[2 + half];
        oacc[t] = __builtin_amdgcn_wmma_f32_16x16x32_bf16(
            false, pa.v, false, vb.v, (short)0, oacc[t], false, false);
      }
    }
    __syncthreads();
  }

  // write O / l_i
#pragma unroll
  for (int r = 0; r < 8; ++r) {
    int srow = wbase + r + 8 * half;
    float inv = (l_i[r] > 0.f) ? 1.f / l_i[r] : 0.f;
    float* op = O + (size_t)(b * SS + srow) * (NHEADS * HEADD) + h * HEADD;
#pragma unroll
    for (int t = 0; t < 4; ++t) op[t * 16 + l16] = oacc[t][r] * inv;
  }
}

// ---------------------------------------------------------------------------
// Host launcher
// ---------------------------------------------------------------------------
extern "C" void kernel_launch(void* const* d_in, const int* in_sizes, int n_in,
                              void* d_out, int out_size, void* d_ws, size_t ws_size,
                              hipStream_t stream) {
  const float* hs  = (const float*)d_in[0];
  const float* ln1 = (const float*)d_in[1];
  const float* ln2 = (const float*)d_in[2];
  const float* wq  = (const float*)d_in[3];
  const float* wk  = (const float*)d_in[4];
  const float* wv  = (const float*)d_in[5];
  const float* wo  = (const float*)d_in[6];
  const float* wr  = (const float*)d_in[7];
  const float* wg  = (const float*)d_in[8];
  const float* wu  = (const float*)d_in[9];
  const float* wd  = (const float*)d_in[10];

  const size_t TH = (size_t)TOK * HDIM;
  float* w    = (float*)d_ws;
  float* x    = w; w += TH;
  float* xn   = w; w += TH;
  float* qbuf = w; w += (size_t)TOK * NHEADS * HEADD;
  float* kbuf = w; w += (size_t)TOK * NKVH * HEADD;
  float* vbuf = w; w += (size_t)TOK * NKVH * HEADD;
  float* ao   = w; w += (size_t)TOK * NHEADS * HEADD;
  float* gbuf = w; w += (size_t)TOK * FFDIM;
  float* hbuf = w; w += (size_t)TOK * FFDIM;
  float* mo   = w; w += TH;
  float* comb = w; w += (size_t)TOK * ENUM;

  hipMemcpyAsync(x, hs, TH * sizeof(float), hipMemcpyDeviceToDevice, stream);

  const int NQ = NHEADS * HEADD, NKVD = NKVH * HEADD;
  for (int l = 0; l < LNUM; ++l) {
    const float* ln1p = ln1 + (size_t)l * HDIM;
    const float* ln2p = ln2 + (size_t)l * HDIM;
    const float* wqp  = wq + (size_t)l * HDIM * NQ;
    const float* wkp  = wk + (size_t)l * HDIM * NKVD;
    const float* wvp  = wv + (size_t)l * HDIM * NKVD;
    const float* wop  = wo + (size_t)l * NQ * HDIM;
    const float* wrp  = wr + (size_t)l * HDIM * ENUM;

    // --- attention block ---
    rmsnorm_kernel<<<TOK, 256, 0, stream>>>(x, ln1p, xn);
    gemm_wmma_kernel<<<dim3(TOK / 128, NQ / 128), 256, 0, stream>>>(
        xn, HDIM, wqp, NQ, qbuf, NQ, TOK, NQ, HDIM, nullptr, 0, 0, nullptr, 0);
    gemm_wmma_kernel<<<dim3(TOK / 128, NKVD / 128), 256, 0, stream>>>(
        xn, HDIM, wkp, NKVD, kbuf, NKVD, TOK, NKVD, HDIM, nullptr, 0, 0, nullptr, 0);
    gemm_wmma_kernel<<<dim3(TOK / 128, NKVD / 128), 256, 0, stream>>>(
        xn, HDIM, wvp, NKVD, vbuf, NKVD, TOK, NKVD, HDIM, nullptr, 0, 0, nullptr, 0);
    rope_kernel<<<dim3(TOK, NHEADS + NKVH), 32, 0, stream>>>(qbuf, kbuf);
    attn_kernel<<<dim3(SS / 128, BB * NHEADS), 256, 0, stream>>>(qbuf, kbuf, vbuf, ao);
    gemm_wmma_kernel<<<dim3(TOK / 128, HDIM / 128), 256, 0, stream>>>(
        ao, NQ, wop, HDIM, xn, HDIM, TOK, HDIM, NQ, nullptr, 0, 0, nullptr, 0);
    resid_clip_kernel<<<(int)(TH / 256), 256, 0, stream>>>(x, xn, x, (int)TH);

    // --- MoE block ---
    rmsnorm_kernel<<<TOK, 256, 0, stream>>>(x, ln2p, xn);
    router_kernel<<<TOK, 32, 0, stream>>>(xn, wrp, comb);
    for (int e = 0; e < ENUM; ++e) {
      const float* wgp = wg + ((size_t)l * ENUM + e) * HDIM * FFDIM;
      const float* wup = wu + ((size_t)l * ENUM + e) * HDIM * FFDIM;
      const float* wdp = wd + ((size_t)l * ENUM + e) * FFDIM * HDIM;
      // g = xn @ wg[e]
      gemm_wmma_kernel<<<dim3(TOK / 128, FFDIM / 128), 256, 0, stream>>>(
          xn, HDIM, wgp, FFDIM, gbuf, FFDIM, TOK, FFDIM, HDIM,
          nullptr, 0, 0, nullptr, 0);
      // h = silu(g) * (xn @ wu[e])
      gemm_wmma_kernel<<<dim3(TOK / 128, FFDIM / 128), 256, 0, stream>>>(
          xn, HDIM, wup, FFDIM, hbuf, FFDIM, TOK, FFDIM, HDIM,
          nullptr, 0, 0, gbuf, 0);
      // mo (+)= comb[:,e] * (h @ wd[e])
      gemm_wmma_kernel<<<dim3(TOK / 128, HDIM / 128), 256, 0, stream>>>(
          hbuf, FFDIM, wdp, HDIM, mo, HDIM, TOK, HDIM, FFDIM,
          comb, ENUM, e, nullptr, (e > 0) ? 1 : 0);
    }
    resid_clip_kernel<<<(int)(TH / 256), 256, 0, stream>>>(x, mo, x, (int)TH);
  }

  hipMemcpyAsync(d_out, x, TH * sizeof(float), hipMemcpyDeviceToDevice, stream);
}
